// Long_Layer_2302102470698
// MI455X (gfx1250) — matrix operand order repaired and checked by reading, not compiled, not run on hardware
//
#include <hip/hip_runtime.h>

#define BM 128
#define BN 128
#define BK 32
#define LDSS 40   // 32 + 8 bf16 pad -> 80B row stride, conflict-free b128 frag loads

typedef __attribute__((ext_vector_type(16))) __bf16 bf16x16;
typedef __attribute__((ext_vector_type(8)))  float  f32x8;

union FragU { uint4 q[2]; bf16x16 v; };

__device__ __forceinline__ unsigned short f32_to_bf16_rne(float f) {
  unsigned int u = __float_as_uint(f);
  u += 0x7FFFu + ((u >> 16) & 1u);
  return (unsigned short)(u >> 16);
}

// ---- one-shot fp32 -> bf16 conversion (weights / input), float4 wide ----
__global__ __launch_bounds__(256) void cvt_f32_bf16(
    const float* __restrict__ s, unsigned short* __restrict__ d, int n4)
{
  const int i = blockIdx.x * blockDim.x + threadIdx.x;
  if (i >= n4) return;
  const float4 f = reinterpret_cast<const float4*>(s)[i];
  ushort4 o;
  o.x = f32_to_bf16_rne(f.x); o.y = f32_to_bf16_rne(f.y);
  o.z = f32_to_bf16_rne(f.z); o.w = f32_to_bf16_rne(f.w);
  reinterpret_cast<ushort4*>(d)[i] = o;
}

// stage one 32B A-slice + one 32B B-slice global -> LDS as a single clause of
// four async b128 copies (ASYNCcnt += 4 per wave); INST_OFFSET hits both sides
__device__ __forceinline__ void async_stage(const unsigned short* gA,
                                            const unsigned short* gB,
                                            unsigned short* lA,
                                            unsigned short* lB) {
  const unsigned int la = (unsigned int)(unsigned long long)lA;
  const unsigned int lb = (unsigned int)(unsigned long long)lB;
  asm volatile(
      "s_clause 0x3\n\t"
      "global_load_async_to_lds_b128 %0, %2, off\n\t"
      "global_load_async_to_lds_b128 %0, %2, off offset:16\n\t"
      "global_load_async_to_lds_b128 %1, %3, off\n\t"
      "global_load_async_to_lds_b128 %1, %3, off offset:16"
      :: "v"(la), "v"(lb), "v"(gA), "v"(gB) : "memory");
}

// C[M,N] = A[M,K(lda)] * W[N,K(ldw)]^T (+bias, opt ReLU); bf16 in, bf16 WMMA,
// f32 accumulate; writes f32 (Cf) and/or bf16 (Cb).
// Double-buffered LDS with async-to-LDS staging; k-loop unrolled x2 so both
// buffer indices are compile-time constants (K/BK is even for every call).
__global__ __launch_bounds__(256) void gemm_bf16_wmma(
    const unsigned short* __restrict__ A, int lda,
    const unsigned short* __restrict__ W, int ldw,
    float* __restrict__ Cf, unsigned short* __restrict__ Cb, int ldc,
    int M, int N, int K,
    const float* __restrict__ bias, int act)
{
  __shared__ unsigned short sA[2][BM * LDSS];
  __shared__ unsigned short sB[2][BN * LDSS];

  const int tid   = threadIdx.x;
  const int lane  = tid & 31;
  const int wid   = tid >> 5;
  const int waveM = wid >> 1;          // 0..3 -> 32-row strip
  const int waveN = wid & 1;           // 0..1 -> 64-col strip
  const int half  = lane >> 4;
  const int l15   = lane & 15;

  const int blockN0 = blockIdx.x * BN;
  const int blockM0 = blockIdx.y * BM;

  const f32x8 zero = {0.f,0.f,0.f,0.f,0.f,0.f,0.f,0.f};
  f32x8 acc[2][4];
  #pragma unroll
  for (int i = 0; i < 2; i++)
    #pragma unroll
    for (int j = 0; j < 4; j++) acc[i][j] = zero;

  // staging: each of 256 threads copies 16 bf16 of A and 16 bf16 of W per tile
  const int srow = tid >> 1;           // 0..127
  const int scol = (tid & 1) * 16;     // 0 / 16
  const unsigned short* gA = A + (size_t)(blockM0 + srow) * lda + scol;
  const unsigned short* gW = W + (size_t)(blockN0 + srow) * ldw + scol;
  unsigned short* lA0 = &sA[0][srow*LDSS + scol];
  unsigned short* lA1 = &sA[1][srow*LDSS + scol];
  unsigned short* lB0 = &sB[0][srow*LDSS + scol];
  unsigned short* lB1 = &sB[1][srow*LDSS + scol];

  const int fa0 = (waveM*32 + l15) * LDSS + half*8;   // A frag base (i stride 16*LDSS)
  const int fb0 = (waveN*64 + l15) * LDSS + half*8;   // B frag base (j stride 16*LDSS)

  // compute one k-tile out of LDS buffer `buf` (compile-time constant)
  #define COMPUTE_TILE(buf)                                                    \
    {                                                                          \
      FragU a_[2], b_[4];                                                      \
      _Pragma("unroll")                                                        \
      for (int i = 0; i < 2; i++) {                                            \
        const unsigned short* p = &sA[buf][fa0 + i*16*LDSS];                   \
        a_[i].q[0] = *reinterpret_cast<const uint4*>(p);                       \
        a_[i].q[1] = *reinterpret_cast<const uint4*>(p + 16);                  \
      }                                                                        \
      _Pragma("unroll")                                                        \
      for (int j = 0; j < 4; j++) {                                            \
        const unsigned short* p = &sB[buf][fb0 + j*16*LDSS];                   \
        b_[j].q[0] = *reinterpret_cast<const uint4*>(p);                       \
        b_[j].q[1] = *reinterpret_cast<const uint4*>(p + 16);                  \
      }                                                                        \
      _Pragma("unroll")                                                        \
      for (int i = 0; i < 2; i++)                                              \
        _Pragma("unroll")                                                      \
        for (int j = 0; j < 4; j++)                                            \
          acc[i][j] = __builtin_amdgcn_wmma_f32_16x16x32_bf16(                 \
              false, a_[i].v, false, b_[j].v, (short)0, acc[i][j], false, false); \
    }

  // prologue: stage k-tile 0 into buffer 0
  async_stage(gA, gW, lA0, lB0);

  for (int k0 = 0; k0 < K; k0 += 2*BK) {
    // ---- even step: compute buf0, prefetch k0+BK into buf1 (always exists) ----
    async_stage(gA + k0 + BK, gW + k0 + BK, lA1, lB1);
    asm volatile("s_wait_asynccnt 4" ::: "memory");   // buf0's 4 copies done
    __syncthreads();
    COMPUTE_TILE(0)
    __syncthreads();

    // ---- odd step: compute buf1, prefetch k0+2BK into buf0 (unless last) ----
    const bool last = (k0 + 2*BK) >= K;
    if (!last) {
      async_stage(gA + k0 + 2*BK, gW + k0 + 2*BK, lA0, lB0);
      asm volatile("s_wait_asynccnt 4" ::: "memory");
    } else {
      asm volatile("s_wait_asynccnt 0" ::: "memory");
    }
    __syncthreads();
    COMPUTE_TILE(1)
    __syncthreads();
  }
  #undef COMPUTE_TILE

  // epilogue: C/D layout lanes0-15 M=v, lanes16-31 M=v+8; N=lane&15
  const int mbase = blockM0 + waveM*32 + half*8;
  const int col0  = blockN0 + waveN*64 + l15;
  #pragma unroll
  for (int j = 0; j < 4; j++) {
    const int col = col0 + j*16;
    if (col < N) {
      const float bv = bias ? bias[col] : 0.f;
      #pragma unroll
      for (int i = 0; i < 2; i++) {
        const int row = mbase + i*16;
        #pragma unroll
        for (int v = 0; v < 8; v++) {
          float val = acc[i][j][v] + bv;
          if (act) val = fmaxf(val, 0.f);
          const size_t off = (size_t)(row + v) * ldc + col;
          if (Cf) Cf[off] = val;
          if (Cb) Cb[off] = f32_to_bf16_rne(val);
        }
      }
    }
  }
}

// causal depthwise conv (DCONV=4) + SiLU; reads u = xz[:, :, :2048],
// writes f32 (for the scan) and bf16 (for the x_proj GEMM)
__global__ __launch_bounds__(256) void conv_silu_kernel(
    const float* __restrict__ xz, const float* __restrict__ cw,
    const float* __restrict__ cb, float* __restrict__ uc,
    unsigned short* __restrict__ ucb)
{
  const int idx = blockIdx.x * blockDim.x + threadIdx.x; // B*L*E = 2^24
  if (idx >= (4 << 22)) return;
  const int e = idx & 2047;
  const int t = (idx >> 11) & 2047;
  const int b = idx >> 22;
  float s = cb[e];
  #pragma unroll
  for (int k = 0; k < 4; k++) {
    const int tt = t - 3 + k;
    if (tt >= 0) s += cw[e*4 + k] * xz[(size_t)(b*2048 + tt) * 4096 + e];
  }
  const float sig = 1.f / (1.f + __expf(-s));
  const float v   = s * sig;
  const size_t o  = (size_t)(b*2048 + t) * 2048 + e;
  uc[o]  = v;
  ucb[o] = f32_to_bf16_rne(v);
}

// selective scan, one thread per (b,e) channel, 16-state in registers,
// fused softplus(dt)+bias, D skip, SiLU(z) gate; bf16 output for out_proj.
__global__ __launch_bounds__(256) void scan_kernel(
    const float* __restrict__ uc, const float* __restrict__ dtraw,
    const float* __restrict__ xdbl, const float* __restrict__ xz,
    const float* __restrict__ A_log, const float* __restrict__ dt_b,
    const float* __restrict__ Dp, unsigned short* __restrict__ ygb)
{
  const int tid = blockIdx.x * blockDim.x + threadIdx.x;
  if (tid >= 4 * 2048) return;
  const int b = tid >> 11;
  const int e = tid & 2047;

  float Arow[16], h[16];
  #pragma unroll
  for (int n = 0; n < 16; n++) { Arow[n] = -__expf(A_log[e*16 + n]); h[n] = 0.f; }
  const float dtb = dt_b[e];
  const float Dv  = Dp[e];

  for (int t = 0; t < 2048; t++) {
    const size_t r = (size_t)(b*2048 + t);
    const float u  = uc[r*2048 + e];
    const float xr = dtraw[r*2048 + e] + dtb;
    const float dt = (xr > 20.f) ? xr : log1pf(__expf(xr));
    const float z  = xz[r*4096 + 2048 + e];
    const float du = dt * u;
    const float4* bc = reinterpret_cast<const float4*>(&xdbl[r*96 + 64]);
    float4 Bq[2] = { bc[0], bc[1] };
    float4 Cq[2] = { bc[2], bc[3] };
    const float* Bv = reinterpret_cast<const float*>(Bq);
    const float* Cv = reinterpret_cast<const float*>(Cq);
    float y = 0.f;
    #pragma unroll
    for (int n = 0; n < 16; n++) {
      h[n] = __expf(dt * Arow[n]) * h[n] + du * Bv[n];
      y += h[n] * Cv[n];
    }
    const float sz = z / (1.f + __expf(-z));
    ygb[r*2048 + e] = f32_to_bf16_rne((y + u * Dv) * sz);
  }
}

extern "C" void kernel_launch(void* const* d_in, const int* in_sizes, int n_in,
                              void* d_out, int out_size, void* d_ws, size_t ws_size,
                              hipStream_t stream) {
  const float* x        = (const float*)d_in[0];
  const float* in_proj  = (const float*)d_in[1];
  const float* conv_w   = (const float*)d_in[2];
  const float* conv_b   = (const float*)d_in[3];
  const float* x_proj   = (const float*)d_in[4];
  const float* dt_proj  = (const float*)d_in[5];
  const float* dt_pb    = (const float*)d_in[6];
  const float* A_log    = (const float*)d_in[7];
  const float* Dp       = (const float*)d_in[8];
  const float* out_proj = (const float*)d_in[9];
  const float* lin1_w   = (const float*)d_in[10];
  const float* lin1_b   = (const float*)d_in[11];
  const float* lin2_w   = (const float*)d_in[12];
  const float* lin2_b   = (const float*)d_in[13];
  float* outp = (float*)d_out;

  char* ws = (char*)d_ws;
  // f32 intermediates
  float* xz    = (float*)(ws);                       // 8192x4096
  float* uc    = (float*)(ws + 134217728ull);        // 8192x2048
  float* xdbl  = (float*)(ws + 201326592ull);        // 8192x96
  float* dtraw = (float*)(ws + 204472320ull);        // 8192x2048
  // bf16 intermediates
  unsigned short* xb    = (unsigned short*)(ws + 271581184ull); // 8192x1024
  unsigned short* ucb   = (unsigned short*)(ws + 288358400ull); // 8192x2048
  unsigned short* xdblb = (unsigned short*)(ws + 321912832ull); // 8192x96
  unsigned short* y1b   = (unsigned short*)(ws + 323485696ull); // 8192x1024
  // bf16 weights
  unsigned short* in_projb = (unsigned short*)(ws + 340262912ull); // 4096x1024
  unsigned short* x_projb  = (unsigned short*)(ws + 348651520ull); // 128x2048 (padded)
  unsigned short* dt_projb = (unsigned short*)(ws + 349175808ull); // 2048x64
  unsigned short* out_projb= (unsigned short*)(ws + 349437952ull); // 1024x2048
  unsigned short* lin1b    = (unsigned short*)(ws + 353632256ull); // 4096x1024
  unsigned short* lin2b    = (unsigned short*)(ws + 362020864ull); // 1024x4096
  // reuse: ucb free after x_proj GEMM -> ygb; xz free after scan -> hffb
  unsigned short* ygb  = ucb;                        // 8192x2048
  unsigned short* hffb = (unsigned short*)(ws);      // 8192x4096 (in xz region)

  const dim3 blk(256);
  const int Mb = 8192 / BM; // 64

  // 0. one-shot bf16 conversions (input + weights)
  cvt_f32_bf16<<<2097152/256, blk, 0, stream>>>(x,        xb,        2097152);
  cvt_f32_bf16<<<1048576/256, blk, 0, stream>>>(in_proj,  in_projb,  1048576);
  cvt_f32_bf16<<<  49152/256, blk, 0, stream>>>(x_proj,   x_projb,     49152);
  cvt_f32_bf16<<<  32768/256, blk, 0, stream>>>(dt_proj,  dt_projb,    32768);
  cvt_f32_bf16<<< 524288/256, blk, 0, stream>>>(out_proj, out_projb,  524288);
  cvt_f32_bf16<<<1048576/256, blk, 0, stream>>>(lin1_w,   lin1b,     1048576);
  cvt_f32_bf16<<<1048576/256, blk, 0, stream>>>(lin2_w,   lin2b,     1048576);

  // 1. xz = x @ in_proj^T                      [8192,4096] f32
  gemm_bf16_wmma<<<dim3(4096/BN, Mb), blk, 0, stream>>>(
      xb, 1024, in_projb, 1024, xz, nullptr, 4096, 8192, 4096, 1024, nullptr, 0);
  // 2. uc = silu(causal_dwconv(u))             [8192,2048] f32 + bf16
  conv_silu_kernel<<<(4 << 22) / 256, blk, 0, stream>>>(xz, conv_w, conv_b, uc, ucb);
  // 3. xdbl = uc @ x_proj^T                    [8192,96] f32 + bf16
  gemm_bf16_wmma<<<dim3(1, Mb), blk, 0, stream>>>(
      ucb, 2048, x_projb, 2048, xdbl, xdblb, 96, 8192, 96, 2048, nullptr, 0);
  // 4. dtraw = xdbl[:, :64] @ dt_proj^T        [8192,2048] f32
  gemm_bf16_wmma<<<dim3(2048/BN, Mb), blk, 0, stream>>>(
      xdblb, 96, dt_projb, 64, dtraw, nullptr, 2048, 8192, 2048, 64, nullptr, 0);
  // 5. selective scan + D skip + SiLU(z) gate  [8192,2048] bf16
  scan_kernel<<<8192 / 256, blk, 0, stream>>>(uc, dtraw, xdbl, xz,
                                              A_log, dt_pb, Dp, ygb);
  // 6. y1 = yg @ out_proj^T                    [8192,1024] bf16
  gemm_bf16_wmma<<<dim3(1024/BN, Mb), blk, 0, stream>>>(
      ygb, 2048, out_projb, 2048, nullptr, y1b, 1024, 8192, 1024, 2048, nullptr, 0);
  // 7. hff = relu(y1 @ lin1^T + b1)            [8192,4096] bf16
  gemm_bf16_wmma<<<dim3(4096/BN, Mb), blk, 0, stream>>>(
      y1b, 1024, lin1b, 1024, nullptr, hffb, 4096, 8192, 4096, 1024, lin1_b, 1);
  // 8. out = hff @ lin2^T + b2                 [8192,1024] f32
  gemm_bf16_wmma<<<dim3(1024/BN, Mb), blk, 0, stream>>>(
      hffb, 4096, lin2b, 4096, outp, nullptr, 1024, 8192, 1024, 4096, lin2_b, 0);
}
